// NBVHModel_73031623901467
// MI455X (gfx1250) — compile-verified, matrix operand order checked by reading, not run on hardware
//
#include <hip/hip_runtime.h>
#include <hip/hip_bf16.h>
#include <stdint.h>
#include <stddef.h>

// ---------------------------------------------------------------------------
// NBVH iteration on MI455X (gfx1250): bf16 WMMA MLP, wave32, WGP-resident LDS
// ---------------------------------------------------------------------------

typedef __attribute__((ext_vector_type(16))) __bf16 v16bf;
typedef __attribute__((ext_vector_type(8)))  __bf16 v8bf;
typedef __attribute__((ext_vector_type(8)))  float  v8f;

#define N_RAYS   65536
#define N_POINTS 32
#define FEAT     128
#define DIM      512
#define K0       4096           // N_POINTS * FEAT
#define N_LAYERS 6
#define TILE_M   32             // rays per workgroup (2 row-tiles of 16)
#define LN_EPS   1e-5f

// ---- dynamic LDS layout (bytes) -------------------------------------------
// phase <=2 : wbuf  f32 [32*32*8]   @ 0        (32 KB)
//             fbuf  bf16[32*1024]   @ 32 KB    (64 KB)
// phase >=3 : accb  f32 [32*512]    @ 0        (64 KB)  (aliases wbuf/fbuf)
// always    : ibuf  bf16[32*512]    @ 96 KB    (32 KB)
//             misc                  @ 128 KB   (~4 KB)
#define OFF_WBUF 0
#define OFF_FBUF (32 * 1024)
#define OFF_ACC  0
#define OFF_IBUF (96 * 1024)
#define OFF_MISC (128 * 1024)
#define SMEM_BYTES (128 * 1024 + 4096)

// Load a 16-element bf16 WMMA fragment: element j holds
// K = (j>>3)*16 + half*8 + (j&7); p points at (row, kbase + half*8) so the
// two halves are two contiguous 16-byte loads at p and p+16 elements.
__device__ __forceinline__ v16bf ld_frag(const __bf16* p) {
    union { v16bf v; v8bf h[2]; } u;
    u.h[0] = *(const v8bf*)(p);
    u.h[1] = *(const v8bf*)(p + 16);
    return u.v;
}

__device__ __forceinline__ v8f vzero8() {
    v8f z = {0.f, 0.f, 0.f, 0.f, 0.f, 0.f, 0.f, 0.f};
    return z;
}

// Scatter one wave's 8 accumulator tiles (+bias) into the fp32 LDS buffer.
// C/D layout: VGPR r, lanes 0-15 -> M=r, N=lane ; lanes 16-31 -> M=r+8.
__device__ __forceinline__ void store_acc(const v8f* acc, float* accb,
                                          const float* __restrict__ bias,
                                          int rt, int cg, int lrow, int half) {
#pragma unroll
    for (int j = 0; j < 8; ++j) {
        const int col = (cg * 8 + j) * 16 + lrow;
        const float bv = bias[col];
#pragma unroll
        for (int r = 0; r < 8; ++r) {
            const int m = rt * 16 + half * 8 + r;
            accb[m * DIM + col] = acc[j][r] + bv;
        }
    }
}

__global__ void convert_bf16_kernel(const float* __restrict__ src,
                                    unsigned short* __restrict__ dstu, int n) {
    __bf16* dst = reinterpret_cast<__bf16*>(dstu);
    int i = blockIdx.x * blockDim.x + threadIdx.x;
    if (i < n) dst[i] = (__bf16)src[i];
}

__global__ __launch_bounds__(256)
void nbvh_kernel(const float* __restrict__ orig,
                 const float* __restrict__ vec,
                 const float* __restrict__ t1g,
                 const float* __restrict__ t2g,
                 const float* __restrict__ dist_in,
                 const float* __restrict__ nodes_min,
                 const float* __restrict__ nodes_max,
                 const float* __restrict__ bbox_features,
                 const unsigned short* __restrict__ W0u,
                 const float* __restrict__ b0,
                 const unsigned short* __restrict__ Wlu,
                 const float* __restrict__ bl,
                 const float* __restrict__ gl,
                 const float* __restrict__ betal,
                 const float* __restrict__ Wcls,
                 const float* __restrict__ bcls,
                 const float* __restrict__ Wdist,
                 const float* __restrict__ bdist,
                 const int* __restrict__ bbox_idxs,
                 const unsigned char* __restrict__ mask,
                 float* __restrict__ out) {
    extern __shared__ char smem[];
    const __bf16* W0b = reinterpret_cast<const __bf16*>(W0u);
    const __bf16* Wlb = reinterpret_cast<const __bf16*>(Wlu);

    float*  wbuf = (float*)(smem + OFF_WBUF);
    __bf16* fbuf = (__bf16*)(smem + OFF_FBUF);
    float*  accb = (float*)(smem + OFF_ACC);
    __bf16* ibuf = (__bf16*)(smem + OFF_IBUF);

    float* oebuf = (float*)(smem + OFF_MISC); // [32][6] o.xyz, e.xyz
    float* lenb  = oebuf + 192;               // [32]
    float* t1b   = lenb + 32;                 // [32]
    float* dinb  = t1b + 32;                  // [32]
    float* red1  = dinb + 32;                 // [32*8]
    float* red2  = red1 + 256;                // [32*8]
    float* mub   = red2 + 256;                // [32]
    float* rsb   = mub + 32;                  // [32]
    int*   idxb  = (int*)(rsb + 32);          // [32]
    int*   mskb  = idxb + 32;                 // [32]

    const int t    = threadIdx.x;
    const int lane = t & 31;
    const int wv   = t >> 5;
    const int lrow = lane & 15;   // M (A) / N (B) row within tile
    const int half = lane >> 4;   // K-half selector
    const int rt   = wv & 1;      // row tile (16 rays each)
    const int cg   = wv >> 1;     // column group (8 N-tiles each)
    const int ray0 = blockIdx.x * TILE_M;

    // ---- phase 1a: per-ray geometry -------------------------------------
    if (t < 32) {
        const int r = ray0 + t;
        const float ox = orig[r * 3 + 0], oy = orig[r * 3 + 1], oz = orig[r * 3 + 2];
        const float vx = vec[r * 3 + 0],  vy = vec[r * 3 + 1],  vz = vec[r * 3 + 2];
        const float t1 = t1g[r], t2 = t2g[r];
        const int   m  = mask[r] ? 1 : 0;
        const int   id = m ? bbox_idxs[r] : 0;
        const float iox = ox + vx * t1, ioy = oy + vy * t1, ioz = oz + vz * t1;
        const float ivx = vx * (t2 - t1), ivy = vy * (t2 - t1), ivz = vz * (t2 - t1);
        const float iex = iox + ivx, iey = ioy + ivy, iez = ioz + ivz;
        const float nx = nodes_min[id * 3 + 0], ny = nodes_min[id * 3 + 1], nz = nodes_min[id * 3 + 2];
        const float dx = nodes_max[id * 3 + 0] - nx;
        const float dy = nodes_max[id * 3 + 1] - ny;
        const float dz = nodes_max[id * 3 + 2] - nz;
        oebuf[t * 6 + 0] = fminf(fmaxf((iox - nx) / dx, 0.f), 1.f);
        oebuf[t * 6 + 1] = fminf(fmaxf((ioy - ny) / dy, 0.f), 1.f);
        oebuf[t * 6 + 2] = fminf(fmaxf((ioz - nz) / dz, 0.f), 1.f);
        oebuf[t * 6 + 3] = fminf(fmaxf((iex - nx) / dx, 0.f), 1.f);
        oebuf[t * 6 + 4] = fminf(fmaxf((iey - ny) / dy, 0.f), 1.f);
        oebuf[t * 6 + 5] = fminf(fmaxf((iez - nz) / dz, 0.f), 1.f);
        lenb[t] = sqrtf(ivx * ivx + ivy * ivy + ivz * ivz);
        t1b[t]  = t1;
        dinb[t] = dist_in[r];
        idxb[t] = id;
        mskb[t] = m;
    }
    __syncthreads();

    // ---- phase 1b: trilinear weights wbuf[n][p][8] ----------------------
#pragma unroll
    for (int q = 0; q < 4; ++q) {
        const int id = t + 256 * q;
        const int n = id >> 5, p = id & 31;
        const float tt = (float)p * (1.0f / 31.0f);
        const float ox = oebuf[n * 6 + 0], oy = oebuf[n * 6 + 1], oz = oebuf[n * 6 + 2];
        const float x = ox + (oebuf[n * 6 + 3] - ox) * tt;
        const float y = oy + (oebuf[n * 6 + 4] - oy) * tt;
        const float z = oz + (oebuf[n * 6 + 5] - oz) * tt;
        const float X = 1.f - x, Y = 1.f - y, Z = 1.f - z;
        float* wp = wbuf + (n * 32 + p) * 8;
        wp[0] = X * Y * Z;  wp[1] = x * Y * Z;  wp[2] = X * y * Z;  wp[3] = X * Y * z;
        wp[4] = x * Y * z;  wp[5] = X * y * z;  wp[6] = x * y * Z;  wp[7] = x * y * z;
    }

    // ---- phase 1c: gather node features into LDS (bf16) -----------------
    {
        const int n = t >> 3, c = t & 7;
        const float* src = bbox_features + (size_t)idxb[n] * 1024 + c * 128;
        __bf16* dst = fbuf + n * 1024 + c * 128;
        for (int f = 0; f < 128; f += 4) {
            const float4 v = *(const float4*)(src + f);
            dst[f + 0] = (__bf16)v.x;
            dst[f + 1] = (__bf16)v.y;
            dst[f + 2] = (__bf16)v.z;
            dst[f + 3] = (__bf16)v.w;
        }
    }
    __syncthreads();

    // ---- phase 2: input GEMM  h0 = interp(32x4096) @ W0^T ---------------
    v8f acc[8];
#pragma unroll
    for (int j = 0; j < 8; ++j) acc[j] = vzero8();

    for (int pg = 0; pg < 8; ++pg) {           // 8 chunks of 4 points (K=512 each)
        {   // build 32x512 bf16 activation chunk
            const int n = t >> 3, fg = t & 7, f0 = fg * 16;
            float wvv[4][8];
#pragma unroll
            for (int pl = 0; pl < 4; ++pl)
#pragma unroll
                for (int c = 0; c < 8; ++c)
                    wvv[pl][c] = wbuf[(n * 32 + pg * 4 + pl) * 8 + c];
#pragma unroll
            for (int fc = 0; fc < 4; ++fc) {
                float fv[8][4];
#pragma unroll
                for (int c = 0; c < 8; ++c)
#pragma unroll
                    for (int u = 0; u < 4; ++u)
                        fv[c][u] = (float)fbuf[n * 1024 + c * 128 + f0 + fc * 4 + u];
#pragma unroll
                for (int pl = 0; pl < 4; ++pl)
#pragma unroll
                    for (int u = 0; u < 4; ++u) {
                        float s = 0.f;
#pragma unroll
                        for (int c = 0; c < 8; ++c) s += wvv[pl][c] * fv[c][u];
                        ibuf[n * DIM + pl * 128 + f0 + fc * 4 + u] = (__bf16)s;
                    }
            }
        }
        __syncthreads();

        const __bf16* arow = ibuf + (rt * 16 + lrow) * DIM + half * 8;
        for (int ks = 0; ks < 16; ++ks) {
            const v16bf a = ld_frag(arow + ks * 32);
            const int kglob = pg * 512 + ks * 32 + half * 8;
#pragma unroll
            for (int j = 0; j < 8; ++j) {
                const __bf16* bp = W0b + (size_t)((cg * 8 + j) * 16 + lrow) * K0 + kglob;
                const v16bf b = ld_frag(bp);
                acc[j] = __builtin_amdgcn_wmma_f32_16x16x32_bf16(
                    false, a, false, b, (short)0, acc[j], false, false);
            }
        }
        __syncthreads();
    }

    // ---- phase 3: writeback h0 + b0 (accb aliases dead wbuf/fbuf) -------
    store_acc(acc, accb, b0, rt, cg, lrow, half);
    __syncthreads();

    // ---- phase 4: six layers: relu -> layernorm -> GEMM -----------------
    for (int L = 0; L < N_LAYERS; ++L) {
        {   // partial sums of relu(x) per ray
            const int n = t >> 3, seg = t & 7;
            float s1 = 0.f, s2 = 0.f;
            const float* row = accb + n * DIM + seg * 64;
            for (int u = 0; u < 64; ++u) {
                const float r = fmaxf(row[u], 0.f);
                s1 += r; s2 += r * r;
            }
            red1[n * 8 + seg] = s1;
            red2[n * 8 + seg] = s2;
        }
        __syncthreads();
        if ((t & 7) == 0) {
            const int n = t >> 3;
            float s1 = 0.f, s2 = 0.f;
            for (int k = 0; k < 8; ++k) { s1 += red1[n * 8 + k]; s2 += red2[n * 8 + k]; }
            const float mu = s1 * (1.f / 512.f);
            const float var = s2 * (1.f / 512.f) - mu * mu;
            mub[n] = mu;
            rsb[n] = rsqrtf(var + LN_EPS);
        }
        __syncthreads();
        {   // normalize + affine -> bf16 activations
            const int n = t >> 3, seg = t & 7;
            const float mu = mub[n], rs = rsb[n];
            const float* g  = gl    + L * DIM;
            const float* be = betal + L * DIM;
            for (int u = 0; u < 64; ++u) {
                const int d = seg * 64 + u;
                const float r = fmaxf(accb[n * DIM + d], 0.f);
                ibuf[n * DIM + d] = (__bf16)((r - mu) * rs * g[d] + be[d]);
            }
        }
        __syncthreads();

        v8f acc2[8];
#pragma unroll
        for (int j = 0; j < 8; ++j) acc2[j] = vzero8();
        const __bf16* Wb = Wlb + (size_t)L * DIM * DIM;
        const __bf16* arow = ibuf + (rt * 16 + lrow) * DIM + half * 8;
        for (int ks = 0; ks < 16; ++ks) {
            const v16bf a = ld_frag(arow + ks * 32);
            const int kg = ks * 32 + half * 8;
#pragma unroll
            for (int j = 0; j < 8; ++j) {
                const __bf16* bp = Wb + (size_t)((cg * 8 + j) * 16 + lrow) * DIM + kg;
                const v16bf b = ld_frag(bp);
                acc2[j] = __builtin_amdgcn_wmma_f32_16x16x32_bf16(
                    false, a, false, b, (short)0, acc2[j], false, false);
            }
        }
        __syncthreads();
        store_acc(acc2, accb, bl + L * DIM, rt, cg, lrow, half);
        __syncthreads();
    }

    // ---- phase 5: final relu + classification / distance heads ----------
    {
        const int n = t >> 3, seg = t & 7;
        float s1 = 0.f, s2 = 0.f;
        for (int u = 0; u < 64; ++u) {
            const int d = seg * 64 + u;
            const float r = fmaxf(accb[n * DIM + d], 0.f);
            s1 += r * Wcls[d];
            s2 += r * Wdist[d];
        }
        red1[n * 8 + seg] = s1;
        red2[n * 8 + seg] = s2;
    }
    __syncthreads();
    if ((t & 7) == 0) {
        const int n = t >> 3;
        const int r = ray0 + n;
        float c = 0.f, dd = 0.f;
        for (int k = 0; k < 8; ++k) { c += red1[n * 8 + k]; dd += red2[n * 8 + k]; }
        const float cls  = c + bcls[0];
        const float dist = (dd + bdist[0]) * lenb[n];
        const int   m    = mskb[n];
        const float hit  = m ? cls : 0.f;
        const float dval = (m ? dist : 0.f) + t1b[n];
        const float din  = dinb[n];
        const bool upd = (hit > 0.f) && (dval < din) && (m != 0);
        out[r]          = hit;
        out[N_RAYS + r] = upd ? dval : din;
    }
}

extern "C" void kernel_launch(void* const* d_in, const int* in_sizes, int n_in,
                              void* d_out, int out_size, void* d_ws, size_t ws_size,
                              hipStream_t stream) {
    const float* orig       = (const float*)d_in[0];
    const float* vecp       = (const float*)d_in[1];
    const float* cur_t1     = (const float*)d_in[2];
    const float* cur_t2     = (const float*)d_in[3];
    const float* dist_in    = (const float*)d_in[4];
    const float* nodes_min  = (const float*)d_in[5];
    const float* nodes_max  = (const float*)d_in[6];
    const float* bbox_feat  = (const float*)d_in[7];
    const float* W0         = (const float*)d_in[8];
    const float* b0         = (const float*)d_in[9];
    const float* Wl         = (const float*)d_in[10];
    const float* bl         = (const float*)d_in[11];
    const float* gl         = (const float*)d_in[12];
    const float* betal      = (const float*)d_in[13];
    const float* Wcls       = (const float*)d_in[14];
    const float* bcls       = (const float*)d_in[15];
    const float* Wdist      = (const float*)d_in[16];
    const float* bdist      = (const float*)d_in[17];
    const int*   bbox_idxs  = (const int*)d_in[18];
    const unsigned char* mask = (const unsigned char*)d_in[19];
    float* out = (float*)d_out;

    // workspace: bf16 copies of the GEMM weights
    unsigned short* W0u = (unsigned short*)d_ws;                 // 512*4096
    unsigned short* Wlu = W0u + (size_t)DIM * K0;                // 6*512*512

    const int n0 = DIM * K0;
    const int n1 = N_LAYERS * DIM * DIM;
    convert_bf16_kernel<<<(n0 + 255) / 256, 256, 0, stream>>>(W0, W0u, n0);
    convert_bf16_kernel<<<(n1 + 255) / 256, 256, 0, stream>>>(Wl, Wlu, n1);

    dim3 grid(N_RAYS / TILE_M);
    nbvh_kernel<<<grid, 256, SMEM_BYTES, stream>>>(
        orig, vecp, cur_t1, cur_t2, dist_in, nodes_min, nodes_max, bbox_feat,
        W0u, b0, Wlu, bl, gl, betal, Wcls, bcls, Wdist, bdist,
        bbox_idxs, mask, out);
}
